// GAT_2757369004091
// MI455X (gfx1250) — matrix operand order repaired and checked by reading, not compiled, not run on hardware
//
#include <hip/hip_runtime.h>
#include <hip/hip_bf16.h>

#define NN 8192
#define K_IN 256
#define D_OUT 128
#define LOG2E 1.4426950408889634f

typedef __attribute__((ext_vector_type(16))) __bf16 v16bf;
typedef __attribute__((ext_vector_type(8)))  float  v8f;

union Frag16 { v16bf v; unsigned int u[8]; };

__device__ __forceinline__ unsigned int bf16rne(float x) {
    unsigned int u = __float_as_uint(x);
    u += 0x7FFFu + ((u >> 16) & 1u);
    return u >> 16;
}

// Kernel 1: h = x@W (per row), write hT in bf16 (transposed, 128 x 8192),
// f1 = h@a[:128], f2 = h@a[128:]. One block = one node row, 128 threads.
__global__ void __launch_bounds__(128)
gat_hproj(const float* __restrict__ x, const float* __restrict__ W,
          const float* __restrict__ a, unsigned short* __restrict__ hT,
          float* __restrict__ f1, float* __restrict__ f2) {
    __shared__ float xrow[K_IN];
    __shared__ float red[D_OUT];
    const int r = blockIdx.x;
    const int t = threadIdx.x;              // 0..127 -> output feature
    xrow[t]       = x[(size_t)r * K_IN + t];
    xrow[t + 128] = x[(size_t)r * K_IN + 128 + t];
    __syncthreads();
    float acc = 0.f;
#pragma unroll 8
    for (int k = 0; k < K_IN; ++k)
        acc = fmaf(xrow[k], W[k * D_OUT + t], acc);
    hT[(size_t)t * NN + r] = (unsigned short)bf16rne(acc);
    red[t] = acc * a[t];
    __syncthreads();
    for (int s = 64; s > 0; s >>= 1) { if (t < s) red[t] += red[t + s]; __syncthreads(); }
    if (t == 0) f1[r] = red[0];
    __syncthreads();
    red[t] = acc * a[D_OUT + t];
    __syncthreads();
    for (int s = 64; s > 0; s >>= 1) { if (t < s) red[t] += red[t + s]; __syncthreads(); }
    if (t == 0) f2[r] = red[0];
}

// Kernel 2: global max of f2 (softmax shift ingredient).
__global__ void __launch_bounds__(256)
gat_fmax(const float* __restrict__ f2, float* __restrict__ fmax) {
    __shared__ float red[256];
    const int t = threadIdx.x;
    float m = -3.0e38f;
    for (int i = t; i < NN; i += 256) m = fmaxf(m, f2[i]);
    red[t] = m;
    __syncthreads();
    for (int s = 128; s > 0; s >>= 1) { if (t < s) red[t] = fmaxf(red[t], red[t + s]); __syncthreads(); }
    if (t == 0) *fmax = red[0];
}

// Kernel 3: fused masked-softmax + P@h via bf16 WMMA, single pass over adj.
// Block = 4 waves; each wave owns 32 rows (2 A fragments) x a 2048-column
// strip.  Fixed per-row shift M_i = lrelu(f1_i + max f2) makes all partials
// purely additive (no online rescaling).  Row sums come from an extra WMMA
// against an all-ones B fragment, so the epilogue needs no shuffles.
__global__ void __launch_bounds__(128)
gat_attn(const int* __restrict__ adj, const unsigned short* __restrict__ hT,
         const float* __restrict__ f1, const float* __restrict__ f2,
         const float* __restrict__ fmaxp, float* __restrict__ out) {
    const int lane = threadIdx.x & 31;
    const int wave = threadIdx.x >> 5;       // 0..3 : column-range split
    const int m    = lane & 15;              // A row / B,C column index
    const int half = lane >> 4;
    const int i0   = blockIdx.x * 32;

    const float Fmax = *fmaxp;
    // exp2-domain constants: p = 2^( C0*t + C1*|t| + Cy ),  t = f1_i + f2_j,
    // C0*t + C1*|t| = lrelu(t)*log2e  (slope 0.5),  Cy = -M_i*log2e
    const float C0 = 0.75f * LOG2E;
    const float C1 = 0.25f * LOG2E;
    const float f1m0 = f1[i0 + m];
    const float f1m1 = f1[i0 + 16 + m];
    float tb0 = f1m0 + Fmax; tb0 = tb0 > 0.f ? tb0 : 0.5f * tb0;
    float tb1 = f1m1 + Fmax; tb1 = tb1 > 0.f ? tb1 : 0.5f * tb1;
    const float Cy0 = -tb0 * LOG2E;
    const float Cy1 = -tb1 * LOG2E;

    v8f acc[2][8];
    v8f accS[2];
#pragma unroll
    for (int g = 0; g < 2; ++g) {
        accS[g] = (v8f)(0.0f);
#pragma unroll
        for (int nt = 0; nt < 8; ++nt) acc[g][nt] = (v8f)(0.0f);
    }
    Frag16 Bone;
#pragma unroll
    for (int v = 0; v < 8; ++v) Bone.u[v] = 0x3F803F80u;   // bf16 1.0 pair

    const size_t adjRow0 = (size_t)(i0 + m) * NN;
    const size_t adjRow1 = adjRow0 + (size_t)16 * NN;
    const int jbeg = wave * (NN / 4);
    const int jend = jbeg + (NN / 4);

    for (int jb = jbeg; jb < jend; jb += 32) {
        // deep prefetch of the one true HBM stream (speculative => OOB-safe)
        __builtin_prefetch(adj + adjRow0 + jb + 1024, 0, 0);
        __builtin_prefetch(adj + adjRow1 + jb + 1024, 0, 0);

        Frag16 A0, A1;
#pragma unroll
        for (int w = 0; w < 2; ++w) {
            // A-layout k range for this VGPR quad: kb..kb+7 (contiguous)
            const int j = jb + 16 * half + 8 * w;
            const float4 fa  = *(const float4*)(f2 + j);
            const float4 fb  = *(const float4*)(f2 + j + 4);
            const int4   r0a = *(const int4*)(adj + adjRow0 + j);
            const int4   r0b = *(const int4*)(adj + adjRow0 + j + 4);
            const int4   r1a = *(const int4*)(adj + adjRow1 + j);
            const int4   r1b = *(const int4*)(adj + adjRow1 + j + 4);
            const float ff[8] = {fa.x, fa.y, fa.z, fa.w, fb.x, fb.y, fb.z, fb.w};
            const int   e0[8] = {r0a.x, r0a.y, r0a.z, r0a.w, r0b.x, r0b.y, r0b.z, r0b.w};
            const int   e1[8] = {r1a.x, r1a.y, r1a.z, r1a.w, r1b.x, r1b.y, r1b.z, r1b.w};
#pragma unroll
            for (int c = 0; c < 4; ++c) {
                const float t0 = f1m0 + ff[2 * c], t1 = f1m0 + ff[2 * c + 1];
                const float t2 = f1m1 + ff[2 * c], t3 = f1m1 + ff[2 * c + 1];
                float y0 = fmaf(C0, t0, fmaf(C1, fabsf(t0), Cy0));
                float y1 = fmaf(C0, t1, fmaf(C1, fabsf(t1), Cy0));
                float y2 = fmaf(C0, t2, fmaf(C1, fabsf(t2), Cy1));
                float y3 = fmaf(C0, t3, fmaf(C1, fabsf(t3), Cy1));
                y0 = e0[2 * c]     > 0 ? y0 : -1000.f;   // exp2(-1000)==0, branchless
                y1 = e0[2 * c + 1] > 0 ? y1 : -1000.f;
                y2 = e1[2 * c]     > 0 ? y2 : -1000.f;
                y3 = e1[2 * c + 1] > 0 ? y3 : -1000.f;
                const unsigned int p0 = __float_as_uint(__builtin_amdgcn_exp2f(y0));
                const unsigned int p1 = __float_as_uint(__builtin_amdgcn_exp2f(y1));
                const unsigned int p2 = __float_as_uint(__builtin_amdgcn_exp2f(y2));
                const unsigned int p3 = __float_as_uint(__builtin_amdgcn_exp2f(y3));
                // one v_perm_b32: pack the two f32 high halves as packed bf16
                A0.u[w * 4 + c] = __builtin_amdgcn_perm(p1, p0, 0x07060302u);
                A1.u[w * 4 + c] = __builtin_amdgcn_perm(p3, p2, 0x07060302u);
            }
        }
#pragma unroll
        for (int nt = 0; nt < 8; ++nt) {
            Frag16 B;
            const unsigned short* hrow =
                hT + (size_t)(nt * 16 + m) * NN + jb + 16 * half;
#pragma unroll
            for (int v = 0; v < 8; ++v)
                B.u[v] = *(const unsigned int*)(hrow + 2 * v);
            acc[0][nt] = __builtin_amdgcn_wmma_f32_16x16x32_bf16(
                false, A0.v, false, B.v, (short)0, acc[0][nt], false, false);
            acc[1][nt] = __builtin_amdgcn_wmma_f32_16x16x32_bf16(
                false, A1.v, false, B.v, (short)0, acc[1][nt], false, false);
        }
        // row-sum WMMA: P @ ones -> softmax denominators, no VALU adds
        accS[0] = __builtin_amdgcn_wmma_f32_16x16x32_bf16(
            false, A0.v, false, Bone.v, (short)0, accS[0], false, false);
        accS[1] = __builtin_amdgcn_wmma_f32_16x16x32_bf16(
            false, A1.v, false, Bone.v, (short)0, accS[1], false, false);
    }

    // ---- combine the 4 column-split waves (pure addition), then normalize
    __shared__ float pacc[3][32][132];
    __shared__ float psum[3][32];
    if (wave != 0) {
#pragma unroll
        for (int g = 0; g < 2; ++g) {
#pragma unroll
            for (int nt = 0; nt < 8; ++nt)
#pragma unroll
                for (int r = 0; r < 8; ++r)
                    pacc[wave - 1][g * 16 + r + 8 * half][nt * 16 + m] = acc[g][nt][r];
#pragma unroll
            for (int r = 0; r < 8; ++r)
                psum[wave - 1][g * 16 + r + 8 * half] = accS[g][r];  // same value all n lanes
        }
    }
    __syncthreads();
    if (wave == 0) {
#pragma unroll
        for (int g = 0; g < 2; ++g) {
#pragma unroll
            for (int r = 0; r < 8; ++r) {
                const int row = g * 16 + r + 8 * half;
                const float rs = accS[g][r] + psum[0][row] + psum[1][row] + psum[2][row];
                const float inv = 1.0f / fmaxf(rs, 1e-30f);
#pragma unroll
                for (int nt = 0; nt < 8; ++nt) {
                    const int col = nt * 16 + m;
                    float val = acc[g][nt][r]
                              + pacc[0][row][col] + pacc[1][row][col] + pacc[2][row][col];
                    val *= inv;
                    val = val > 0.f ? val : expm1f(val);     // ELU (alpha=1)
                    out[(size_t)(i0 + row) * D_OUT + col] = val;
                }
            }
        }
    }
}

extern "C" void kernel_launch(void* const* d_in, const int* in_sizes, int n_in,
                              void* d_out, int out_size, void* d_ws, size_t ws_size,
                              hipStream_t stream) {
    const float* x   = (const float*)d_in[0];   // 8192 x 256
    const int*   adj = (const int*)d_in[1];     // 8192 x 8192
    const float* W   = (const float*)d_in[2];   // 256 x 128
    const float* a   = (const float*)d_in[3];   // 256
    float* out = (float*)d_out;                 // 8192 x 128

    unsigned short* hT = (unsigned short*)d_ws;                       // 2 MB bf16 h^T
    float* f1   = (float*)((char*)d_ws + (size_t)D_OUT * NN * 2);
    float* f2   = f1 + NN;
    float* fmax = f2 + NN;

    gat_hproj<<<NN, 128, 0, stream>>>(x, W, a, hT, f1, f2);
    gat_fmax<<<1, 256, 0, stream>>>(f2, fmax);
    gat_attn<<<NN / 32, 128, 0, stream>>>(adj, hT, f1, f2, fmax, out);
}
